// MultiHeadSelfAttentionANE_66228395705292
// MI455X (gfx1250) — matrix operand order repaired and checked by reading, not compiled, not run on hardware
//
#include <hip/hip_runtime.h>
#include <hip/hip_bf16.h>

// Problem constants (match reference)
constexpr int BN = 4;      // batch
constexpr int DN = 1024;   // model dim
constexpr int LN = 2048;   // sequence length
constexpr int HN = 16;     // heads
constexpr int DH = 64;     // head dim
// SCALE = sqrt(DH) = 8  -> multiply scores by 0.125f

typedef __bf16 v16bf __attribute__((ext_vector_type(16)));
typedef float  v8f   __attribute__((ext_vector_type(8)));

union FragBF {
    v16bf v;
    unsigned int u[8];
    __bf16 h[16];
};

struct __align__(16) BF8 { __bf16 h[8]; };

// D = A*B + C, A/B bf16 16x16x32, C/D f32
__device__ __forceinline__ v8f wmma_bf16(v16bf a, v16bf b, v8f c) {
    return __builtin_amdgcn_wmma_f32_16x16x32_bf16(
        /*neg_a=*/false, a, /*neg_b=*/false, b,
        /*c_mod=*/(short)0, c, /*reuse_a=*/false, /*reuse_b=*/false);
}

// Column mapping for a 16x32 A tile in "fragment order":
// a lane with kb=(lane>>4)*16 needs K = {kb..kb+15}; identity mapping keeps
// those 16 elements contiguous at offset kb.
__device__ __forceinline__ int apack32(int k) {
    return k;
}
__device__ __forceinline__ int apack64(int k) {  // two independent 32-wide K slices
    return (k & 32) | apack32(k & 31);
}

// Contiguous 16-element fragment read (works for B tiles stored [n][k] and for
// A tiles stored with apack column permutation).
//   row = row0 + (lane&15) ; element e at base[row*pitch + k0 + (lane>>4)*16 + e]
// -> two 16B LDS reads per lane (compiler merges the u32s into ds_load_b128).
__device__ __forceinline__ v16bf load_frag_contig16(const __bf16* base, int row0, int k0, int pitch) {
    const int lane = threadIdx.x & 31;
    const int row  = row0 + (lane & 15);
    const int kb   = k0 + (lane >> 4) * 16;
    const __bf16* rp = base + row * pitch + kb;
    FragBF f;
#pragma unroll
    for (int p = 0; p < 8; ++p) f.u[p] = *(const unsigned int*)(rp + 2 * p);
    return f.v;
}

// A fragment gathered from global memory where element (m,K) lives at g[K*stride + m]
// (used for Q^T: Q stored (Dh, L), we need (query, Dh) fragments)
__device__ __forceinline__ v16bf load_frag_a_gstride(const __bf16* g, long long stride) {
    const int lane = threadIdx.x & 31;
    const int m  = lane & 15;
    const int kb = (lane >> 4) * 16;
    FragBF f;
#pragma unroll
    for (int e = 0; e < 16; ++e) {
        const int K = kb + e;   // kb..kb+15
        f.h[e] = g[(long long)K * stride + m];
    }
    return f.v;
}

// intra-16-lane-half reductions (C-layout rows are per-VGPR, shared by a 16-lane half)
__device__ __forceinline__ float half_max(float x) {
#pragma unroll
    for (int m = 1; m < 16; m <<= 1) x = fmaxf(x, __shfl_xor(x, m, 32));
    return x;
}
__device__ __forceinline__ float half_sum(float x) {
#pragma unroll
    for (int m = 1; m < 16; m <<= 1) x += __shfl_xor(x, m, 32);
    return x;
}

// ---------------- pack f32 -> bf16 (8 elements / thread, b128 traffic) ----------------
__global__ void pack_bf16_kernel(const float* __restrict__ in, __bf16* __restrict__ out, int n8) {
    const int i = blockIdx.x * blockDim.x + threadIdx.x;
    if (i < n8) {
        const float4* in4 = (const float4*)in;
        const float4 a = in4[2 * i];
        const float4 b = in4[2 * i + 1];
        BF8 o;
        o.h[0] = (__bf16)a.x; o.h[1] = (__bf16)a.y; o.h[2] = (__bf16)a.z; o.h[3] = (__bf16)a.w;
        o.h[4] = (__bf16)b.x; o.h[5] = (__bf16)b.y; o.h[6] = (__bf16)b.z; o.h[7] = (__bf16)b.w;
        *(BF8*)(out + (size_t)i * 8) = o;
    }
}

// ---------------- GEMM:  Y[b] = W (D x D) * X[b] (D x L) ----------------
// 64(M) x 128(N) block tile, 128 threads = 4 waves; each wave does 32x64
// (2x4 WMMA tiles): 8 WMMA per 6 fragment loads per K step.
template <int OUT_F32>
__global__ __launch_bounds__(128) void gemm_wx_kernel(
        const __bf16* __restrict__ W, const __bf16* __restrict__ X,
        __bf16* __restrict__ Yb, float* __restrict__ Yf, const float* __restrict__ bias) {
    const int b  = blockIdx.z;
    const int o0 = blockIdx.y * 64;
    const int l0 = blockIdx.x * 128;
    const __bf16* Xb = X + (size_t)b * DN * LN;

    const int t    = threadIdx.x;
    const int wave = t >> 5;
    const int lane = t & 31;
    const int wr = wave >> 1, wc = wave & 1;    // 2(M) x 2(N) wave grid
    const int half = lane >> 4, nl = lane & 15;

    __shared__ __bf16 sA[64 * 40];    // [m][apack(k)]  (W tile, fragment-order columns)
    __shared__ __bf16 sB[128 * 40];   // [n][k]         (X tile, transposed)

    v8f acc[2][4] = {};

    for (int k0 = 0; k0 < DN; k0 += 32) {
        // stage A: W[o0+m][k0+k] -> sA[m*40 + apack(k)]  (global reads coalesced over k)
#pragma unroll
        for (int i = 0; i < 16; ++i) {
            const int idx = t + i * 128;
            const int m = idx >> 5, k = idx & 31;
            sA[m * 40 + apack32(k)] = W[(size_t)(o0 + m) * DN + k0 + k];
        }
        // stage B transposed: X[k0+k][l0+n] -> sB[n*40+k]  (coalesced over n)
#pragma unroll
        for (int i = 0; i < 32; ++i) {
            const int idx = t + i * 128;
            const int n = idx & 127, k = idx >> 7;
            sB[n * 40 + k] = Xb[(size_t)(k0 + k) * LN + l0 + n];
        }
        if (k0 + 32 < DN) {  // hint next K tile (global_prefetch_b8)
            __builtin_prefetch(W + (size_t)(o0 + (t >> 1)) * DN + k0 + 32, 0, 1);
            __builtin_prefetch(Xb + (size_t)(k0 + 32 + (t >> 2)) * LN + l0, 0, 1);
        }
        __syncthreads();

        v16bf af[2], bf[4];
#pragma unroll
        for (int mt = 0; mt < 2; ++mt) af[mt] = load_frag_contig16(sA, wr * 32 + mt * 16, 0, 40);
#pragma unroll
        for (int nt = 0; nt < 4; ++nt) bf[nt] = load_frag_contig16(sB, wc * 64 + nt * 16, 0, 40);
#pragma unroll
        for (int mt = 0; mt < 2; ++mt)
#pragma unroll
            for (int nt = 0; nt < 4; ++nt)
                acc[mt][nt] = wmma_bf16(af[mt], bf[nt], acc[mt][nt]);

        __syncthreads();
    }

    // store: C layout -> row m = r + 8*half, col n = nl
#pragma unroll
    for (int mt = 0; mt < 2; ++mt)
#pragma unroll
        for (int nt = 0; nt < 4; ++nt) {
            const int obase = o0 + wr * 32 + mt * 16 + half * 8;
            const int l = l0 + wc * 64 + nt * 16 + nl;
#pragma unroll
            for (int r = 0; r < 8; ++r) {
                const int o = obase + r;
                const float v = acc[mt][nt][r];
                if (OUT_F32)
                    Yf[(size_t)b * DN * LN + (size_t)o * LN + l] = v + bias[o];
                else
                    Yb[(size_t)b * DN * LN + (size_t)o * LN + l] = (__bf16)v;
            }
        }
}

// ---------------- flash attention ----------------
// grid (L/64, H, B), 128 threads = 4 waves; each wave owns 16 queries.
__global__ __launch_bounds__(128) void attn_kernel(
        const __bf16* __restrict__ Q, const __bf16* __restrict__ K,
        const __bf16* __restrict__ V, __bf16* __restrict__ O) {
    const int b  = blockIdx.z;
    const int h  = blockIdx.y;
    const int i0 = blockIdx.x * 64;

    const int t = threadIdx.x, wave = t >> 5, lane = t & 31;
    const int half = lane >> 4, nl = lane & 15;

    const size_t bo = (size_t)b * DN * LN;
    const __bf16* Qh = Q + bo + (size_t)(h * DH) * LN;
    const __bf16* Kh = K + bo + (size_t)(h * DH) * LN;
    const __bf16* Vh = V + bo + (size_t)(h * DH) * LN;

    __shared__ __bf16 sK[64 * 72];        // [key][dh]        (K^T tile)
    __shared__ __bf16 sV[64 * 72];        // [dh][key]        (V tile, natural layout)
    __shared__ __bf16 sP[4][16 * 72];     // per-wave P: [i][apack64(j)] bf16 (fragment order)

    // Q^T A-fragments: rows = 16 queries of this wave, K dim = Dh (2 K-tiles of 32)
    v16bf aq0, aq1;
    {
        const __bf16* qb = Qh + (i0 + wave * 16);   // element (m,K) at qb[K*LN + m]
        aq0 = load_frag_a_gstride(qb, LN);
        aq1 = load_frag_a_gstride(qb + (size_t)32 * LN, LN);
    }

    v8f oacc[4] = {};                 // 16 queries x 64 dh accumulator
    float rmax[8], rsum[8];
#pragma unroll
    for (int r = 0; r < 8; ++r) { rmax[r] = -3.0e38f; rsum[r] = 0.f; }

    for (int j0 = 0; j0 < LN; j0 += 64) {
        // stage K^T: sK[n*72 + k] = Kh[k*LN + j0 + n]   (coalesced over keys n)
#pragma unroll
        for (int i = 0; i < 32; ++i) {
            const int idx = t + i * 128;
            const int n = idx & 63, k = idx >> 6;
            sK[n * 72 + k] = Kh[(size_t)k * LN + j0 + n];
        }
        // stage V: sV[d*72 + j] = Vh[d*LN + j0 + j]     (coalesced over keys j)
#pragma unroll
        for (int i = 0; i < 32; ++i) {
            const int idx = t + i * 128;
            const int j = idx & 63, d = idx >> 6;
            sV[d * 72 + j] = Vh[(size_t)d * LN + j0 + j];
        }
        __syncthreads();

        // S = (Q^T K) / 8 : 4 N-tiles of 16 keys, K dim = Dh = 2x32
        v8f s[4];
#pragma unroll
        for (int nt = 0; nt < 4; ++nt) {
            v8f z = {};
            const v16bf bk0 = load_frag_contig16(sK, nt * 16, 0, 72);
            const v16bf bk1 = load_frag_contig16(sK, nt * 16, 32, 72);
            z = wmma_bf16(aq0, bk0, z);
            z = wmma_bf16(aq1, bk1, z);
            s[nt] = z * 0.125f;
        }

        // online softmax, per output row r (row m = r + 8*half, shared by 16-lane half)
        float corr[8];
#pragma unroll
        for (int r = 0; r < 8; ++r) {
            float mx = fmaxf(fmaxf(s[0][r], s[1][r]), fmaxf(s[2][r], s[3][r]));
            mx = half_max(mx);
            const float mnew = fmaxf(rmax[r], mx);
            const float c = __expf(rmax[r] - mnew);
            float psum = 0.f;
#pragma unroll
            for (int nt = 0; nt < 4; ++nt) {
                const float pv = __expf(s[nt][r] - mnew);
                s[nt][r] = pv;
                psum += pv;
            }
            psum = half_sum(psum);
            rsum[r] = rsum[r] * c + psum;
            rmax[r] = mnew;
            corr[r] = c;
        }
#pragma unroll
        for (int nt = 0; nt < 4; ++nt)
#pragma unroll
            for (int r = 0; r < 8; ++r) oacc[nt][r] *= corr[r];

        // P (C-layout f32) -> LDS bf16 in A-fragment column order
        __bf16* pw = sP[wave];
#pragma unroll
        for (int nt = 0; nt < 4; ++nt)
#pragma unroll
            for (int r = 0; r < 8; ++r)
                pw[(r + 8 * half) * 72 + apack64(nt * 16 + nl)] = (__bf16)s[nt][r];

        const v16bf ap0 = load_frag_contig16(pw, 0, 0, 72);   // keys 0..31
        const v16bf ap1 = load_frag_contig16(pw, 0, 32, 72);  // keys 32..63

        // O += P * V^T : N = dh (4 tiles of 16), K = 64 keys (2 K-tiles of 32)
#pragma unroll
        for (int nt = 0; nt < 4; ++nt) {
            const v16bf bv0 = load_frag_contig16(sV, nt * 16, 0, 72);
            const v16bf bv1 = load_frag_contig16(sV, nt * 16, 32, 72);
            oacc[nt] = wmma_bf16(ap0, bv0, oacc[nt]);
            oacc[nt] = wmma_bf16(ap1, bv1, oacc[nt]);
        }
        __syncthreads();
    }

    // normalize + store to (B, D, L) bf16: row d = h*64 + nt*16 + nl, cols contiguous in r
    float inv[8];
#pragma unroll
    for (int r = 0; r < 8; ++r) inv[r] = 1.f / (rsum[r] + 1e-30f);

    __bf16* Ob = O + bo;
#pragma unroll
    for (int nt = 0; nt < 4; ++nt) {
        const int d = h * DH + nt * 16 + nl;
        const int lbase = i0 + wave * 16 + half * 8;
        BF8 outv;
#pragma unroll
        for (int r = 0; r < 8; ++r) outv.h[r] = (__bf16)(oacc[nt][r] * inv[r]);
        *(BF8*)(Ob + (size_t)d * LN + lbase) = outv;
    }
}

// ---------------- host launcher ----------------
extern "C" void kernel_launch(void* const* d_in, const int* in_sizes, int n_in,
                              void* d_out, int out_size, void* d_ws, size_t ws_size,
                              hipStream_t stream) {
    const float* x  = (const float*)d_in[0];   // (B, D, 1, L) == (B, D, L)
    const float* wq = (const float*)d_in[1];
    const float* wk = (const float*)d_in[2];
    const float* wv = (const float*)d_in[3];
    const float* wo = (const float*)d_in[4];
    const float* bo = (const float*)d_in[5];
    float* out = (float*)d_out;

    const size_t nW = (size_t)DN * DN;        // 1M
    const size_t nX = (size_t)BN * DN * LN;   // 8M

    char* p = (char*)d_ws;
    auto take = [&](size_t n) { __bf16* r = (__bf16*)p; p += n * sizeof(__bf16); return r; };
    __bf16* xb  = take(nX);
    __bf16* wqb = take(nW);
    __bf16* wkb = take(nW);
    __bf16* wvb = take(nW);
    __bf16* wob = take(nW);
    __bf16* qb  = take(nX);
    __bf16* kb  = take(nX);
    __bf16* vb  = take(nX);
    __bf16* ab  = take(nX);   // attention output (B, D, L)

    // 1) pack fp32 -> bf16 (vectorized: 8 elems / thread)
    pack_bf16_kernel<<<dim3((unsigned)((nX / 8 + 255) / 256)), 256, 0, stream>>>(x,  xb,  (int)(nX / 8));
    pack_bf16_kernel<<<dim3((unsigned)((nW / 8 + 255) / 256)), 256, 0, stream>>>(wq, wqb, (int)(nW / 8));
    pack_bf16_kernel<<<dim3((unsigned)((nW / 8 + 255) / 256)), 256, 0, stream>>>(wk, wkb, (int)(nW / 8));
    pack_bf16_kernel<<<dim3((unsigned)((nW / 8 + 255) / 256)), 256, 0, stream>>>(wv, wvb, (int)(nW / 8));
    pack_bf16_kernel<<<dim3((unsigned)((nW / 8 + 255) / 256)), 256, 0, stream>>>(wo, wob, (int)(nW / 8));

    // 2) QKV projections
    dim3 gg(LN / 128, DN / 64, BN);
    gemm_wx_kernel<0><<<gg, 128, 0, stream>>>(wqb, xb, qb, nullptr, nullptr);
    gemm_wx_kernel<0><<<gg, 128, 0, stream>>>(wkb, xb, kb, nullptr, nullptr);
    gemm_wx_kernel<0><<<gg, 128, 0, stream>>>(wvb, xb, vb, nullptr, nullptr);

    // 3) flash attention per (b, h, 64-query block)
    attn_kernel<<<dim3(LN / 64, HN, BN), 128, 0, stream>>>(qb, kb, vb, ab);

    // 4) output projection + bias -> f32 out
    gemm_wx_kernel<1><<<gg, 128, 0, stream>>>(wob, ab, nullptr, out, bo);
}
